// Transformer_14886356648887
// MI455X (gfx1250) — matrix operand order repaired and checked by reading, not compiled, not run on hardware
//
#include <hip/hip_runtime.h>

// ---------------- problem constants (from reference) ----------------
#define M_B   256              // batch
#define S_Q   128              // sequence
#define C_CH  256              // channels
#define N_HD  8                // heads
#define D_H   32               // head dim
#define M_S   (M_B * S_Q)      // 32768 rows for all projections
#define LDT   40               // LDS tile row stride in halfs (80B, 16B aligned)
#define LDV   136              // LDS v-transpose row stride in halfs (272B, 16B aligned)

typedef __attribute__((ext_vector_type(16))) _Float16 v16h;
typedef __attribute__((ext_vector_type(8)))  float    v8f;

union Frag16 { v16h v; uint4 q[2]; };
union H8     { uint4 q; _Float16 h[8]; };

// Load a 32-byte WMMA fragment as two 16B LDS reads (-> ds_load_b128).
__device__ __forceinline__ v16h frag_ld(const _Float16* lo, const _Float16* hi) {
  Frag16 f;
  f.q[0] = *(const uint4*)lo;
  f.q[1] = *(const uint4*)hi;
  return f.v;
}

// CDNA5 async global->LDS copy (GLOBAL_LOAD_ASYNC_TO_LDS_B128, ASYNCcnt).
// Generic LDS pointer low 32 bits are the wave-relative LDS byte address.
__device__ __forceinline__ void async_cp16(_Float16* lds_dst, const _Float16* g_src) {
  unsigned           ldsa = (unsigned)(unsigned long long)(void*)lds_dst;
  unsigned long long ga   = (unsigned long long)(const void*)g_src;
  asm volatile("global_load_async_to_lds_b128 %0, %1, off"
               :: "v"(ldsa), "v"(ga) : "memory");
}
__device__ __forceinline__ void async_wait0() {
  asm volatile("s_wait_asynccnt 0x0" ::: "memory");
}

// ---------------------------------------------------------------
// Generic f16 GEMM:  out[M,N] = A[M,K] * W[N,K]^T (+bias)(+resid)
// Block tile 128x64, 8 waves (4x2), wave tile 32x32 (2x2 WMMA tiles).
// Double-buffered LDS; tiles fetched with async copies overlapping WMMA.
// One barrier per K step.
// ---------------------------------------------------------------
__global__ __launch_bounds__(256) void gemm_f16_kernel(
    const _Float16* __restrict__ A, const _Float16* __restrict__ W,
    const float* __restrict__ bias, const float* __restrict__ resid,
    float* __restrict__ outF, _Float16* __restrict__ outH,
    int M, int N, int K)
{
  __shared__ _Float16 As[2][128 * LDT];
  __shared__ _Float16 Bs[2][64 * LDT];
  const int t    = threadIdx.x;
  const int lane = t & 31;
  const int wv   = t >> 5;
  const int hi   = lane >> 4;
  const int l15  = lane & 15;
  const int row0 = blockIdx.x * 128;
  const int col0 = blockIdx.y * 64;
  const int m0   = (wv & 3) * 32;
  const int n0   = (wv >> 2) * 32;

  v8f acc00 = {}, acc01 = {}, acc10 = {}, acc11 = {};

  const int r  = t >> 2;          // 0..63
  const int cb = (t & 3) * 8;     // half offset within 32-wide K tile

  auto issue_tiles = [&](int k0, int b) {
    async_cp16(&As[b][r * LDT + cb],        A + (size_t)(row0 + r) * K + k0 + cb);
    async_cp16(&As[b][(r + 64) * LDT + cb], A + (size_t)(row0 + r + 64) * K + k0 + cb);
    async_cp16(&Bs[b][r * LDT + cb],        W + (size_t)(col0 + r) * K + k0 + cb);
  };

  issue_tiles(0, 0);
  int cur = 0;
  for (int k0 = 0; k0 < K; k0 += 32) {
    async_wait0();                 // my copies into buf[cur] landed
    __syncthreads();               // everyone's landed; everyone's prior reads done
    if (k0 + 32 < K) issue_tiles(k0 + 32, cur ^ 1);   // overlap with compute

    const _Float16* pa0 = As[cur] + (m0 + l15) * LDT;
    const _Float16* pa1 = As[cur] + (m0 + 16 + l15) * LDT;
    const _Float16* pb0 = Bs[cur] + (n0 + l15) * LDT + hi * 16;
    const _Float16* pb1 = Bs[cur] + (n0 + 16 + l15) * LDT + hi * 16;
    v16h a0 = frag_ld(pa0 + hi * 16, pa0 + hi * 16 + 8);
    v16h a1 = frag_ld(pa1 + hi * 16, pa1 + hi * 16 + 8);
    v16h b0 = frag_ld(pb0, pb0 + 8);
    v16h b1 = frag_ld(pb1, pb1 + 8);
    acc00 = __builtin_amdgcn_wmma_f32_16x16x32_f16(false, a0, false, b0, (short)0, acc00, false, false);
    acc01 = __builtin_amdgcn_wmma_f32_16x16x32_f16(false, a0, false, b1, (short)0, acc01, false, false);
    acc10 = __builtin_amdgcn_wmma_f32_16x16x32_f16(false, a1, false, b0, (short)0, acc10, false, false);
    acc11 = __builtin_amdgcn_wmma_f32_16x16x32_f16(false, a1, false, b1, (short)0, acc11, false, false);
    cur ^= 1;
  }

  v8f* accs[4] = { &acc00, &acc01, &acc10, &acc11 };
  for (int idx = 0; idx < 4; ++idx) {
    const int mi = idx >> 1, ni = idx & 1;
    for (int rr = 0; rr < 8; ++rr) {
      const int row = row0 + m0 + mi * 16 + hi * 8 + rr;   // D layout: M = 8*hi + r
      const int col = col0 + n0 + ni * 16 + l15;           // N = lane&15
      float vv = (*accs[idx])[rr];
      if (bias)  vv += bias[col];
      if (resid) vv += resid[(size_t)row * N + col];
      if (outF) outF[(size_t)row * N + col] = vv;
      else      outH[(size_t)row * N + col] = (_Float16)vv;
    }
  }
}

// ---------------------------------------------------------------
// Batched attention scores: one block per (n,h); energy = q_nh * k_nh^T
// q/k tiles are 128x32 (d=32 contiguous per row), single WMMA K step.
// Waves 2x4, wave tile 64x32 -> 8 WMMAs per wave. Async tile fetch.
// ---------------------------------------------------------------
__global__ __launch_bounds__(256) void attn_energy_kernel(
    const _Float16* __restrict__ q, const _Float16* __restrict__ k,
    float* __restrict__ energy)
{
  __shared__ _Float16 Qs[128 * LDT];
  __shared__ _Float16 Ks[128 * LDT];
  const int t    = threadIdx.x;
  const int lane = t & 31;
  const int wv   = t >> 5;
  const int hi   = lane >> 4;
  const int l15  = lane & 15;
  const int nh   = blockIdx.x;
  const int nb   = nh / N_HD;
  const int hd   = nh % N_HD;
  const size_t gbase = ((size_t)nb * S_Q) * C_CH + (size_t)hd * D_H;

  const int r  = t >> 2;
  const int cb = (t & 3) * 8;
  async_cp16(&Qs[r * LDT + cb],        q + gbase + (size_t)r * C_CH + cb);
  async_cp16(&Qs[(r + 64) * LDT + cb], q + gbase + (size_t)(r + 64) * C_CH + cb);
  async_cp16(&Ks[r * LDT + cb],        k + gbase + (size_t)r * C_CH + cb);
  async_cp16(&Ks[(r + 64) * LDT + cb], k + gbase + (size_t)(r + 64) * C_CH + cb);
  async_wait0();
  __syncthreads();

  const int m0 = (wv & 1) * 64;
  const int n0 = (wv >> 1) * 32;

  const _Float16* pb0 = Ks + (n0 + l15) * LDT + hi * 16;
  const _Float16* pb1 = Ks + (n0 + 16 + l15) * LDT + hi * 16;
  v16h b0 = frag_ld(pb0, pb0 + 8);
  v16h b1 = frag_ld(pb1, pb1 + 8);

  for (int mi = 0; mi < 4; ++mi) {
    const _Float16* pa = Qs + (m0 + mi * 16 + l15) * LDT;
    v16h a = frag_ld(pa + hi * 16, pa + hi * 16 + 8);
    v8f acc0 = {}, acc1 = {};
    acc0 = __builtin_amdgcn_wmma_f32_16x16x32_f16(false, a, false, b0, (short)0, acc0, false, false);
    acc1 = __builtin_amdgcn_wmma_f32_16x16x32_f16(false, a, false, b1, (short)0, acc1, false, false);
    for (int rr = 0; rr < 8; ++rr) {
      const int row = m0 + mi * 16 + hi * 8 + rr;
      energy[((size_t)nh * S_Q + row) * S_Q + n0 + l15]      = acc0[rr];
      energy[((size_t)nh * S_Q + row) * S_Q + n0 + 16 + l15] = acc1[rr];
    }
  }
}

// ---------------------------------------------------------------
// Batched attn @ v: one block per (n,h); o_nh[128x32] = attn[128x128] * v[128x32].
// V transposed into LDS once (ds_store scatter); attn K tiles double-buffered
// via async copies; 4 K steps of 32; 8 waves, 16 rows each.
// ---------------------------------------------------------------
__global__ __launch_bounds__(256) void attn_v_kernel(
    const _Float16* __restrict__ attn, const _Float16* __restrict__ v,
    _Float16* __restrict__ o)
{
  __shared__ _Float16 As[2][128 * LDT];
  __shared__ _Float16 Vs[32 * LDV];
  const int t    = threadIdx.x;
  const int lane = t & 31;
  const int wv   = t >> 5;
  const int hi   = lane >> 4;
  const int l15  = lane & 15;
  const int nh   = blockIdx.x;
  const int nb   = nh / N_HD;
  const int hd   = nh % N_HD;
  const size_t vbase = ((size_t)nb * S_Q) * C_CH + (size_t)hd * D_H;

  const int r  = t >> 2;
  const int cb = (t & 3) * 8;
  for (int it = 0; it < 2; ++it) {         // transpose v_nh -> Vs[c][l]
    const int l = r + it * 64;
    H8 u;
    u.q = *(const uint4*)(v + vbase + (size_t)l * C_CH + cb);
    for (int j = 0; j < 8; ++j) Vs[(cb + j) * LDV + l] = u.h[j];
  }

  auto issue_tile = [&](int k0, int b) {
    async_cp16(&As[b][r * LDT + cb],
               attn + ((size_t)nh * S_Q + r) * S_Q + k0 + cb);
    async_cp16(&As[b][(r + 64) * LDT + cb],
               attn + ((size_t)nh * S_Q + r + 64) * S_Q + k0 + cb);
  };

  const int m0 = wv * 16;
  v8f acc0 = {}, acc1 = {};
  issue_tile(0, 0);
  int cur = 0;
  for (int k0 = 0; k0 < S_Q; k0 += 32) {
    async_wait0();
    __syncthreads();                       // also covers Vs ds_stores (DScnt)
    if (k0 + 32 < S_Q) issue_tile(k0 + 32, cur ^ 1);
    const _Float16* pa  = As[cur] + (m0 + l15) * LDT;
    v16h a = frag_ld(pa + hi * 16, pa + hi * 16 + 8);
    const _Float16* pb0 = Vs + l15 * LDV + k0 + hi * 16;
    const _Float16* pb1 = Vs + (16 + l15) * LDV + k0 + hi * 16;
    v16h b0 = frag_ld(pb0, pb0 + 8);
    v16h b1 = frag_ld(pb1, pb1 + 8);
    acc0 = __builtin_amdgcn_wmma_f32_16x16x32_f16(false, a, false, b0, (short)0, acc0, false, false);
    acc1 = __builtin_amdgcn_wmma_f32_16x16x32_f16(false, a, false, b1, (short)0, acc1, false, false);
    cur ^= 1;
  }
  for (int rr = 0; rr < 8; ++rr) {
    const int row = m0 + hi * 8 + rr;
    o[vbase + (size_t)row * C_CH + l15]      = (_Float16)acc0[rr];
    o[vbase + (size_t)row * C_CH + 16 + l15] = (_Float16)acc1[rr];
  }
}

// ---------------------------------------------------------------
// Deterministic per-column mean / rstd over `rows` samples (biased var).
// One block per column, serial per-thread sums + LDS tree reduce.
// ---------------------------------------------------------------
__global__ __launch_bounds__(256) void col_stats_kernel(
    const float* __restrict__ data, int rows, int cols,
    float* __restrict__ mean, float* __restrict__ rstd)
{
  __shared__ float ssum[256];
  __shared__ float ssq[256];
  const int col = blockIdx.x;
  const int t = threadIdx.x;
  float s = 0.f, s2 = 0.f;
  for (int rr = t; rr < rows; rr += 256) {
    float vv = data[(size_t)rr * cols + col];
    s += vv; s2 += vv * vv;
  }
  ssum[t] = s; ssq[t] = s2;
  __syncthreads();
  for (int off = 128; off > 0; off >>= 1) {
    if (t < off) { ssum[t] += ssum[t + off]; ssq[t] += ssq[t + off]; }
    __syncthreads();
  }
  if (t == 0) {
    const float inv = 1.0f / (float)rows;
    const float mu  = ssum[0] * inv;
    const float var = ssq[0] * inv - mu * mu;
    mean[col] = mu;
    rstd[col] = rsqrtf(var + 1e-5f);
  }
}

// (x - mean)*rstd*g + b [, relu] -> f16
__global__ void norm_affine_kernel(
    const float* __restrict__ in, _Float16* __restrict__ out,
    const float* __restrict__ mean, const float* __restrict__ rstd,
    const float* __restrict__ g, const float* __restrict__ b,
    size_t n, int cols, int relu)
{
  const size_t i = (size_t)blockIdx.x * blockDim.x + threadIdx.x;
  if (i >= n) return;
  const int c = (int)(i % (size_t)cols);
  float vv = (in[i] - mean[c]) * rstd[c] * g[c] + b[c];
  if (relu) vv = fmaxf(vv, 0.f);
  out[i] = (_Float16)vv;
}

// Fused pe-BatchNorm + 1/sqrt(C) scale + softmax over one 128-length row.
__global__ __launch_bounds__(128) void softmax_row_kernel(
    const float* __restrict__ energy,
    const float* __restrict__ mean, const float* __restrict__ rstd,
    const float* __restrict__ g_pe, const float* __restrict__ b_pe,
    _Float16* __restrict__ attn)
{
  __shared__ float red[128];
  const int rowid = blockIdx.x;             // nh*S + qi
  const int qi = rowid & (S_Q - 1);
  const int t = threadIdx.x;
  const int p = qi * S_Q + t;               // position in flattened S*S
  const size_t base = (size_t)rowid * S_Q;
  const float e = energy[base + t];
  const float val = ((e - mean[p]) * rstd[p] * g_pe[p] + b_pe[p]) * 0.0625f; // 1/sqrt(256)
  red[t] = val;
  __syncthreads();
  for (int off = 64; off > 0; off >>= 1) {
    if (t < off) red[t] = fmaxf(red[t], red[t + off]);
    __syncthreads();
  }
  const float mx = red[0];
  __syncthreads();
  const float ex = __expf(val - mx);
  red[t] = ex;
  __syncthreads();
  for (int off = 64; off > 0; off >>= 1) {
    if (t < off) red[t] += red[t + off];
    __syncthreads();
  }
  const float inv = 1.0f / red[0];
  attn[base + t] = (_Float16)(ex * inv);
}

__global__ void cvt_f32_to_f16_kernel(const float* __restrict__ in,
                                      _Float16* __restrict__ out, int n)
{
  const int i = blockIdx.x * blockDim.x + threadIdx.x;
  if (i < n) out[i] = (_Float16)in[i];
}

// ---------------------------------------------------------------
extern "C" void kernel_launch(void* const* d_in, const int* in_sizes, int n_in,
                              void* d_out, int out_size, void* d_ws, size_t ws_size,
                              hipStream_t stream) {
  (void)in_sizes; (void)n_in; (void)out_size; (void)ws_size;
  const float* iX   = (const float*)d_in[0];
  const float* iGn  = (const float*)d_in[1];
  const float* iBn  = (const float*)d_in[2];
  const float* iWq  = (const float*)d_in[3];
  const float* iWk  = (const float*)d_in[4];
  const float* iWv  = (const float*)d_in[5];
  const float* iWo  = (const float*)d_in[6];
  const float* iBo  = (const float*)d_in[7];
  const float* iGpe = (const float*)d_in[8];
  const float* iBpe = (const float*)d_in[9];
  const float* iG0  = (const float*)d_in[10];
  const float* iB0  = (const float*)d_in[11];
  const float* iW0  = (const float*)d_in[12];
  const float* iG1  = (const float*)d_in[13];
  const float* iB1  = (const float*)d_in[14];
  const float* iW1  = (const float*)d_in[15];

  // ---- workspace carve (256B aligned slabs) ----
  char* w = (char*)d_ws;
  auto carve = [&](size_t bytes) -> void* {
    void* p = (void*)w;
    w += (bytes + 255) & ~(size_t)255;
    return p;
  };
  _Float16* Wq16 = (_Float16*)carve((size_t)C_CH * C_CH * 2);
  _Float16* Wk16 = (_Float16*)carve((size_t)C_CH * C_CH * 2);
  _Float16* Wv16 = (_Float16*)carve((size_t)C_CH * C_CH * 2);
  _Float16* Wo16 = (_Float16*)carve((size_t)C_CH * C_CH * 2);
  _Float16* W016 = (_Float16*)carve((size_t)2 * C_CH * C_CH * 2);
  _Float16* W116 = (_Float16*)carve((size_t)2 * C_CH * C_CH * 2);
  float* mean_x  = (float*)carve(C_CH * 4);
  float* rstd_x  = (float*)carve(C_CH * 4);
  float* mean_pe = (float*)carve((size_t)S_Q * S_Q * 4);
  float* rstd_pe = (float*)carve((size_t)S_Q * S_Q * 4);
  float* mean_x1 = (float*)carve(C_CH * 4);
  float* rstd_x1 = (float*)carve(C_CH * 4);
  float* mean_o2 = (float*)carve(2 * C_CH * 4);
  float* rstd_o2 = (float*)carve(2 * C_CH * 4);
  _Float16* h16  = (_Float16*)carve((size_t)M_S * C_CH * 2);
  _Float16* q16  = (_Float16*)carve((size_t)M_S * C_CH * 2);
  _Float16* k16  = (_Float16*)carve((size_t)M_S * C_CH * 2);
  _Float16* v16f = (_Float16*)carve((size_t)M_S * C_CH * 2);
  float*    x1   = (float*)carve((size_t)M_S * C_CH * 4);
  float*    energy = (float*)carve((size_t)M_B * N_HD * S_Q * S_Q * 4);
  _Float16* attn16 = (_Float16*)carve((size_t)M_B * N_HD * S_Q * S_Q * 2);
  // buffer reuse
  _Float16* o16   = h16;      // h dead after QKV
  _Float16* h1_16 = q16;      // q dead after energy
  float*    out2  = energy;   // energy dead after softmax (67MB <= 134MB)
  _Float16* out4  = attn16;   // attn dead after attn@v (33.5MB <= 67MB)

  const size_t nXC = (size_t)M_S * C_CH;

  // 1) weights -> f16
  cvt_f32_to_f16_kernel<<<(C_CH * C_CH + 255) / 256, 256, 0, stream>>>(iWq, Wq16, C_CH * C_CH);
  cvt_f32_to_f16_kernel<<<(C_CH * C_CH + 255) / 256, 256, 0, stream>>>(iWk, Wk16, C_CH * C_CH);
  cvt_f32_to_f16_kernel<<<(C_CH * C_CH + 255) / 256, 256, 0, stream>>>(iWv, Wv16, C_CH * C_CH);
  cvt_f32_to_f16_kernel<<<(C_CH * C_CH + 255) / 256, 256, 0, stream>>>(iWo, Wo16, C_CH * C_CH);
  cvt_f32_to_f16_kernel<<<(2 * C_CH * C_CH + 255) / 256, 256, 0, stream>>>(iW0, W016, 2 * C_CH * C_CH);
  cvt_f32_to_f16_kernel<<<(2 * C_CH * C_CH + 255) / 256, 256, 0, stream>>>(iW1, W116, 2 * C_CH * C_CH);

  // 2) BN(x) -> h (f16)
  col_stats_kernel<<<C_CH, 256, 0, stream>>>(iX, M_S, C_CH, mean_x, rstd_x);
  norm_affine_kernel<<<(unsigned)((nXC + 255) / 256), 256, 0, stream>>>(
      iX, h16, mean_x, rstd_x, iGn, iBn, nXC, C_CH, 0);

  // 3) QKV projections (WMMA)
  dim3 gC(M_S / 128, C_CH / 64);
  gemm_f16_kernel<<<gC, 256, 0, stream>>>(h16, Wq16, nullptr, nullptr, nullptr, q16, M_S, C_CH, C_CH);
  gemm_f16_kernel<<<gC, 256, 0, stream>>>(h16, Wk16, nullptr, nullptr, nullptr, k16, M_S, C_CH, C_CH);
  gemm_f16_kernel<<<gC, 256, 0, stream>>>(h16, Wv16, nullptr, nullptr, nullptr, v16f, M_S, C_CH, C_CH);

  // 4) energy = q k^T per (n,h)   [f32, needed for global pe-BN stats]
  attn_energy_kernel<<<M_B * N_HD, 256, 0, stream>>>(q16, k16, energy);

  // 5) pe-BN stats over (m,H) per S*S position
  col_stats_kernel<<<S_Q * S_Q, 256, 0, stream>>>(energy, M_B * N_HD, S_Q * S_Q, mean_pe, rstd_pe);

  // 6) fused BN + scale + softmax -> attn (f16)
  softmax_row_kernel<<<M_B * N_HD * S_Q, 128, 0, stream>>>(
      energy, mean_pe, rstd_pe, iGpe, iBpe, attn16);

  // 7) o = attn @ v  (WMMA)
  attn_v_kernel<<<M_B * N_HD, 256, 0, stream>>>(attn16, v16f, o16);

  // 8) x1 = o Wo^T + bo + x  (WMMA, fused epilogue)
  gemm_f16_kernel<<<gC, 256, 0, stream>>>(o16, Wo16, iBo, iX, x1, nullptr, M_S, C_CH, C_CH);

  // 9) FFN: BN(x1)+relu -> h1 (f16)
  col_stats_kernel<<<C_CH, 256, 0, stream>>>(x1, M_S, C_CH, mean_x1, rstd_x1);
  norm_affine_kernel<<<(unsigned)((nXC + 255) / 256), 256, 0, stream>>>(
      x1, h1_16, mean_x1, rstd_x1, iG0, iB0, nXC, C_CH, 1);

  // 10) out2 = h1 W0^T  (f32 for BN stats)
  dim3 g2C(M_S / 128, (2 * C_CH) / 64);
  gemm_f16_kernel<<<g2C, 256, 0, stream>>>(h1_16, W016, nullptr, nullptr, out2, nullptr, M_S, 2 * C_CH, C_CH);

  // 11) BN(out2)+relu -> out4 (f16)
  col_stats_kernel<<<2 * C_CH, 256, 0, stream>>>(out2, M_S, 2 * C_CH, mean_o2, rstd_o2);
  norm_affine_kernel<<<(unsigned)((2 * nXC + 255) / 256), 256, 0, stream>>>(
      out2, out4, mean_o2, rstd_o2, iG1, iB1, 2 * nXC, 2 * C_CH, 1);

  // 12) final: d_out = x1 + out4 W1^T  (WMMA, fused residual)
  gemm_f16_kernel<<<gC, 256, 0, stream>>>(out4, W116, nullptr, x1, (float*)d_out, nullptr, M_S, C_CH, 2 * C_CH);
}
